// TFBertSelfAttention_6571299963617
// MI455X (gfx1250) — matrix operand order repaired and checked by reading, not compiled, not run on hardware
//
#include <hip/hip_runtime.h>
#include <hip/hip_bf16.h>

typedef _Float16 half_t;
typedef __attribute__((ext_vector_type(16))) _Float16 v16h;
typedef __attribute__((ext_vector_type(8)))  float    v8f;

#define B_DIM  2
#define S_LEN  2048
#define D_DIM  1024
#define H_NUM  16
#define DH_DIM 64

union FragU { v16h v; uint4 u[2]; };

__device__ inline v8f vzero8() {
  v8f z = {0.f, 0.f, 0.f, 0.f, 0.f, 0.f, 0.f, 0.f};
  return z;
}

// ---------------- CDNA5 async copy helpers (ASYNCcnt-tracked) ----------------
// global_load_async_to_lds_b128: VDST = 32-bit LDS byte address, VADDR = 64-bit
// global address, no SADDR ("off"). Generic->LDS address = low 32 bits.
__device__ __forceinline__ unsigned lds_addr_of(const void* p) {
  return (unsigned)(unsigned long long)p;
}
__device__ __forceinline__ void async_b128(unsigned lds_off, const void* g) {
  asm volatile("global_load_async_to_lds_b128 %0, %1, off"
               :: "v"(lds_off), "v"((unsigned long long)g)
               : "memory");
}
#define WAIT_ASYNCCNT(n) asm volatile("s_wait_asynccnt %0" ::"n"(n) : "memory")

// A fragment (16x32 f16): lane = row (lane&15), regs = K with 8-wide lane-group split.
__device__ inline v16h ld_fragA(const half_t* base, int strideH, int kbaseH) {
  int lane = threadIdx.x & 31;
  int r = lane & 15;
  int g = (lane >> 4) & 1;
  const half_t* p = base + (size_t)r * strideH + kbaseH + g * 8;
  FragU f;
  f.u[0] = *(const uint4*)(p);       // VGPRs 0..3  (K = kbase + g*8 .. +7)
  f.u[1] = *(const uint4*)(p + 16);  // VGPRs 4..7  (K = kbase + 16 + g*8 .. +7)
  return f.v;
}

// B fragment (32x16 f16): lane = column N (lane&15), regs = K with 8-wide lane-group split.
__device__ inline v16h ld_fragB(const half_t* base, int strideH, int nbaseH) {
  int lane = threadIdx.x & 31;
  int n = lane & 15;
  int g = (lane >> 4) & 1;
  const half_t* p = base + nbaseH + n;
  FragU f;
#pragma unroll
  for (int s = 0; s < 8; ++s) {
    f.v[s] = p[(size_t)(g * 8 + s) * strideH];
    f.v[8 + s] = p[(size_t)(16 + g * 8 + s) * strideH];
  }
  return f.v;
}

__device__ inline v8f wmma16(v16h a, v16h b, v8f c) {
  return __builtin_amdgcn_wmma_f32_16x16x32_f16(false, a, false, b, (short)0, c,
                                                false, false);
}

// ---------------------------------------------------------------- convert f32 -> f16
__global__ __launch_bounds__(256) void cvt_f16(const float* __restrict__ x,
                                               half_t* __restrict__ y) {
  size_t i = ((size_t)blockIdx.x * 256 + threadIdx.x) * 4;
  float4 f = *(const float4*)(x + i);
  union { half_t h[4]; uint2 u; } pk;
  pk.h[0] = (half_t)f.x; pk.h[1] = (half_t)f.y;
  pk.h[2] = (half_t)f.z; pk.h[3] = (half_t)f.w;
  *(uint2*)(y + i) = pk.u;
}

// ---------------------------------------------------------------- fused QKV projection
// O = X(4096x1024) * W(1024x1024) + b ; block tile 256x128, BK=32, 8 waves,
// each wave computes a 32x128 sub-tile (16 WMMA per K-step). Double-buffered
// async global->LDS staging.
// mode 0: Q  -> [B,H,S,Dh]   mode 1: K -> [B,H,Dh,S]   mode 2: V -> [B,H,S,Dh]
__global__ __launch_bounds__(256) void qkv_gemm(
    const half_t* __restrict__ Xh,
    const half_t* __restrict__ W16q, const half_t* __restrict__ W16k,
    const half_t* __restrict__ W16v,
    const float* __restrict__ bq, const float* __restrict__ bk,
    const float* __restrict__ bv,
    half_t* __restrict__ Qh, half_t* __restrict__ KTh, half_t* __restrict__ Vh) {
  const int mode = blockIdx.z;
  const half_t* W16 = (mode == 0) ? W16q : ((mode == 1) ? W16k : W16v);
  const float* bias = (mode == 0) ? bq : ((mode == 1) ? bk : bv);
  half_t* dst = (mode == 0) ? Qh : ((mode == 1) ? KTh : Vh);

  const int n0 = blockIdx.x * 128;
  const int m0 = blockIdx.y * 256;
  const int tid = threadIdx.x;
  const int wave = tid >> 5;
  const int lane = tid & 31;
  const int r = lane & 15;
  const int g = (lane >> 4) & 1;

  __shared__ half_t Xs[2][256 * 40];  // 256 rows x 32 halves, stride 40
  __shared__ half_t Ws[2][32 * 136];  // 32 k-rows x 128 n-halves, stride 136

  // issue one K-step tile (6 async b128 per thread: 4 X + 2 W)
  auto issue_tile = [&](int kk_t, int buf) {
#pragma unroll
    for (int cc = 0; cc < 4; ++cc) {
      int c = tid + cc * 256;
      int row = c >> 2, part = c & 3;
      async_b128(lds_addr_of(&Xs[buf][row * 40 + part * 8]),
                 Xh + (size_t)(m0 + row) * D_DIM + kk_t + part * 8);
    }
#pragma unroll
    for (int cc = 0; cc < 2; ++cc) {
      int c = tid + cc * 256;
      int row = c >> 4, part = c & 15;
      async_b128(lds_addr_of(&Ws[buf][row * 136 + part * 8]),
                 W16 + (size_t)(kk_t + row) * D_DIM + n0 + part * 8);
    }
  };

  v8f acc[2][8];
#pragma unroll
  for (int u = 0; u < 2; ++u)
#pragma unroll
    for (int i = 0; i < 8; ++i) acc[u][i] = vzero8();

  issue_tile(0, 0);
  for (int ks = 0; ks < D_DIM / 32; ++ks) {
    if (ks + 1 < D_DIM / 32) {
      issue_tile((ks + 1) * 32, (ks + 1) & 1);
      WAIT_ASYNCCNT(6);  // in-order: previous tile's 6 ops have landed
    } else {
      WAIT_ASYNCCNT(0);
    }
    __syncthreads();
    const half_t* Xb = Xs[ks & 1];
    const half_t* Wb = Ws[ks & 1];
    v16h a0 = ld_fragA(Xb + (wave * 32) * 40, 40, 0);
    v16h a1 = ld_fragA(Xb + (wave * 32 + 16) * 40, 40, 0);
#pragma unroll
    for (int nt = 0; nt < 8; ++nt) {
      v16h bfr = ld_fragB(Wb, 136, nt * 16);
      acc[0][nt] = wmma16(a0, bfr, acc[0][nt]);
      acc[1][nt] = wmma16(a1, bfr, acc[1][nt]);
    }
    __syncthreads();
  }

  // epilogue: bias, f16 convert, head split (with transpose for K)
#pragma unroll
  for (int u = 0; u < 2; ++u) {
#pragma unroll
    for (int nt = 0; nt < 8; ++nt) {
      int n = n0 + nt * 16 + r;
      float bval = bias[n];
      int h = n >> 6, d = n & 63;
#pragma unroll
      for (int v = 0; v < 8; ++v) {
        int row = m0 + wave * 32 + u * 16 + v + 8 * g;  // global row in [0, B*S)
        int b = row >> 11, s = row & 2047;
        float val = acc[u][nt][v] + bval;
        size_t idx;
        if (mode == 1)
          idx = ((size_t)(b * H_NUM + h) * DH_DIM + d) * S_LEN + s;
        else
          idx = ((size_t)(b * H_NUM + h) * S_LEN + s) * DH_DIM + d;
        dst[idx] = (half_t)val;
      }
    }
  }
}

// ---------------------------------------------------------------- flash attention
// Block: 128 threads = 4 waves; each wave owns 16 q rows; block covers 64 q rows.
// 64-key tiles, double-buffered async K/V staging, online softmax.
// 16 WMMA per tile iteration (8 QK^T + 8 P*V), 32 iterations.
__global__ __launch_bounds__(128) void attn(
    const half_t* __restrict__ Qg, const half_t* __restrict__ KTg,
    const half_t* __restrict__ Vg, const float* __restrict__ mask,
    const float* __restrict__ head_mask, float* __restrict__ out) {
  const int qblk = blockIdx.x;
  const int h = blockIdx.y;
  const int b = blockIdx.z;
  const int tid = threadIdx.x;
  const int wave = tid >> 5;
  const int lane = tid & 31;
  const int r = lane & 15;
  const int g = (lane >> 4) & 1;

  __shared__ half_t KTs[2][64 * 72];  // [d=64][keys=64], stride 72 halves
  __shared__ half_t Vs[2][64 * 72];   // [keys=64][d=64], stride 72 halves
  __shared__ half_t Ps[4][16 * 72];   // per-wave P tile [16 q][64 keys]

  const size_t headOff = (size_t)(b * H_NUM + h) * S_LEN * DH_DIM;
  const half_t* Qh = Qg + headOff;
  const half_t* KTh = KTg + headOff;  // [Dh][S]
  const half_t* Vh = Vg + headOff;    // [S][Dh]
  const int q0 = qblk * 64;

  // issue one key tile (8 async b128 per thread: 4 K^T + 4 V)
  auto issue_tile = [&](int kb_t, int buf) {
    int row = tid >> 1;      // 0..63
    int hf = tid & 1;        // half-row (32 halves = 4 x b128)
    const half_t* ks = KTh + (size_t)row * S_LEN + kb_t + hf * 32;
    const half_t* vs = Vh + (size_t)(kb_t + row) * DH_DIM + hf * 32;
    half_t* kd = &KTs[buf][row * 72 + hf * 32];
    half_t* vd = &Vs[buf][row * 72 + hf * 32];
#pragma unroll
    for (int j = 0; j < 4; ++j) {
      async_b128(lds_addr_of(kd + j * 8), ks + j * 8);
      async_b128(lds_addr_of(vd + j * 8), vs + j * 8);
    }
  };

  // Q A-fragments straight from global (row-major per head)
  v16h aq[2];
  {
    const half_t* qrow = Qh + (size_t)(q0 + wave * 16 + r) * DH_DIM;
    FragU f0, f1;
    f0.u[0] = *(const uint4*)(qrow + g * 8);
    f0.u[1] = *(const uint4*)(qrow + 16 + g * 8);
    f1.u[0] = *(const uint4*)(qrow + 32 + g * 8);
    f1.u[1] = *(const uint4*)(qrow + 48 + g * 8);
    aq[0] = f0.v;
    aq[1] = f1.v;
  }

  v8f ctx[4];
#pragma unroll
  for (int i = 0; i < 4; ++i) ctx[i] = vzero8();
  float m_run[8], l_run[8];
#pragma unroll
  for (int v = 0; v < 8; ++v) { m_run[v] = -1e30f; l_run[v] = 0.f; }

  issue_tile(0, 0);
  const int T = S_LEN / 64;  // 32 tiles
  for (int it = 0; it < T; ++it) {
    const int kb = it * 64;
    if (it + 1 < T) {
      issue_tile(kb + 64, (it + 1) & 1);
      WAIT_ASYNCCNT(8);  // previous tile's 8 ops complete (in-order)
    } else {
      WAIT_ASYNCCNT(0);
    }
    __syncthreads();
    const half_t* Kb = KTs[it & 1];
    const half_t* Vb = Vs[it & 1];

    // scores: 4 key-tiles x (2 d-chunks) WMMA
    v8f sc[4];
#pragma unroll
    for (int nt = 0; nt < 4; ++nt) {
      v8f c = vzero8();
      c = wmma16(aq[0], ld_fragB(Kb, 72, nt * 16), c);
      c = wmma16(aq[1], ld_fragB(Kb + 32 * 72, 72, nt * 16), c);
      sc[nt] = c;
    }
    // scale 1/sqrt(64) + additive mask (per key)
#pragma unroll
    for (int nt = 0; nt < 4; ++nt) {
      float mv = mask[(size_t)b * S_LEN + kb + nt * 16 + r];
#pragma unroll
      for (int v = 0; v < 8; ++v) sc[nt][v] = sc[nt][v] * 0.125f + mv;
    }
    // online softmax (row = v + 8g, 16-lane half-wave reductions)
    float alpha[8];
#pragma unroll
    for (int v = 0; v < 8; ++v) {
      float t = sc[0][v];
#pragma unroll
      for (int nt = 1; nt < 4; ++nt) t = fmaxf(t, sc[nt][v]);
#pragma unroll
      for (int off = 8; off >= 1; off >>= 1) t = fmaxf(t, __shfl_xor(t, off, 32));
      float mn = fmaxf(m_run[v], t);
      alpha[v] = __expf(m_run[v] - mn);
      m_run[v] = mn;
    }
#pragma unroll
    for (int v = 0; v < 8; ++v) {
      float ls = 0.f;
#pragma unroll
      for (int nt = 0; nt < 4; ++nt) {
        float p = __expf(sc[nt][v] - m_run[v]);
        sc[nt][v] = p;
        ls += p;
      }
#pragma unroll
      for (int off = 8; off >= 1; off >>= 1) ls += __shfl_xor(ls, off, 32);
      l_run[v] = l_run[v] * alpha[v] + ls;
    }
#pragma unroll
    for (int dt = 0; dt < 4; ++dt)
#pragma unroll
      for (int v = 0; v < 8; ++v) ctx[dt][v] *= alpha[v];

    // C-layout -> A-layout for P via per-wave LDS scratch
    half_t* Pw = Ps[wave];
#pragma unroll
    for (int nt = 0; nt < 4; ++nt)
#pragma unroll
      for (int v = 0; v < 8; ++v)
        Pw[(v + 8 * g) * 72 + nt * 16 + r] = (half_t)sc[nt][v];

    v16h ap[2];
#pragma unroll
    for (int kc = 0; kc < 2; ++kc) ap[kc] = ld_fragA(Pw, 72, kc * 32);
#pragma unroll
    for (int dt = 0; dt < 4; ++dt) {
      v8f c = ctx[dt];
#pragma unroll
      for (int kc = 0; kc < 2; ++kc)
        c = wmma16(ap[kc], ld_fragB(Vb + kc * 32 * 72, 72, dt * 16), c);
      ctx[dt] = c;
    }
    __syncthreads();
  }

  // finalize: /l, *head_mask, merge heads to [B,S,D] f32
  float hm = head_mask[h];
#pragma unroll
  for (int v = 0; v < 8; ++v) {
    float inv = hm / fmaxf(l_run[v], 1e-30f);
    int s = q0 + wave * 16 + v + 8 * g;
    float* orow = out + (size_t)(b * S_LEN + s) * D_DIM + h * DH_DIM;
#pragma unroll
    for (int dt = 0; dt < 4; ++dt) orow[dt * 16 + r] = ctx[dt][v] * inv;
  }
}

// ----------------------------------------------------------------
extern "C" void kernel_launch(void* const* d_in, const int* in_sizes, int n_in,
                              void* d_out, int out_size, void* d_ws, size_t ws_size,
                              hipStream_t stream) {
  (void)in_sizes; (void)n_in; (void)out_size; (void)ws_size;
  const float* hidden = (const float*)d_in[0];
  const float* mask   = (const float*)d_in[1];
  const float* hmask  = (const float*)d_in[2];
  const float* Wq = (const float*)d_in[3];
  const float* bq = (const float*)d_in[4];
  const float* Wk = (const float*)d_in[5];
  const float* bk = (const float*)d_in[6];
  const float* Wv = (const float*)d_in[7];
  const float* bv = (const float*)d_in[8];
  float* out = (float*)d_out;

  char* ws = (char*)d_ws;
  half_t* Xh   = (half_t*)(ws);                          // 8 MiB
  half_t* W16q = (half_t*)(ws + (8u << 20));             // 2 MiB
  half_t* W16k = (half_t*)(ws + (10u << 20));            // 2 MiB
  half_t* W16v = (half_t*)(ws + (12u << 20));            // 2 MiB
  half_t* Qh   = (half_t*)(ws + (14u << 20));            // 8 MiB [B,H,S,Dh]
  half_t* KTh  = (half_t*)(ws + (22u << 20));            // 8 MiB [B,H,Dh,S]
  half_t* Vh   = (half_t*)(ws + (30u << 20));            // 8 MiB [B,H,S,Dh]

  cvt_f16<<<4096, 256, 0, stream>>>(hidden, Xh);  // 4M elems
  cvt_f16<<<1024, 256, 0, stream>>>(Wq, W16q);    // 1M elems each
  cvt_f16<<<1024, 256, 0, stream>>>(Wk, W16k);
  cvt_f16<<<1024, 256, 0, stream>>>(Wv, W16v);

  qkv_gemm<<<dim3(8, 16, 3), 256, 0, stream>>>(Xh, W16q, W16k, W16v, bq, bk, bv,
                                               Qh, KTh, Vh);
  attn<<<dim3(S_LEN / 64, H_NUM, B_DIM), 128, 0, stream>>>(Qh, KTh, Vh, mask,
                                                           hmask, out);
}
